// CVAREncoder_30185030156559
// MI455X (gfx1250) — compile-verified, matrix operand order checked by reading, not run on hardware
//
#include <hip/hip_runtime.h>
#include <math.h>

// ---------------------------------------------------------------------------
// CVAE encoder for MI455X (gfx1250): bf16 WMMA GEMMs + persistent GRU scan.
// Dims: V=32000 E=256 H=512 LAT=128 C=16 NL=2, B=32, L=1024.
// ---------------------------------------------------------------------------

typedef __bf16 v16bf __attribute__((ext_vector_type(16)));
typedef float  v8f   __attribute__((ext_vector_type(8)));

#define BB   32
#define LL   1024
#define HH   512
#define G3H  1536      // 3*H
#define H2   1024      // 2*H
#define MM   (BB*LL)   // 32768 rows
#define SCAN_WGS 32

__device__ __forceinline__ v8f wmma_bf16(v16bf a, v16bf b, v8f c) {
  // D = A(16x32 bf16) x B(32x16 bf16) + C(16x16 f32)
  return __builtin_amdgcn_wmma_f32_16x16x32_bf16(
      /*neg_a=*/false, a, /*neg_b=*/false, b,
      /*c_mod=*/(short)0, c, /*reuse_a=*/false, /*reuse_b=*/false);
}

// ---------------------------------------------------------------------------
// f32 -> bf16 flat conversion
// ---------------------------------------------------------------------------
__global__ void f2bf_kernel(const float* __restrict__ src,
                            __bf16* __restrict__ dst, int n) {
  int i = blockIdx.x * blockDim.x + threadIdx.x;
  if (i < n) dst[i] = (__bf16)src[i];
}

// ---------------------------------------------------------------------------
// Embedding: xs[m,e] = emb[x[m],e] + cemb[c[b],e]  (bf16 out), E=256
// ---------------------------------------------------------------------------
__global__ __launch_bounds__(256)
void embed_kernel(const int* __restrict__ x, const int* __restrict__ c,
                  const float* __restrict__ emb, const float* __restrict__ cemb,
                  __bf16* __restrict__ xs) {
  int m = blockIdx.x;            // 0..32767
  int b = m >> 10;
  int e = threadIdx.x;           // 0..255
  size_t tok = (size_t)x[m];
  size_t cls = (size_t)c[b];
  float v = emb[tok * 256 + e] + cemb[cls * 256 + e];
  xs[(size_t)m * 256 + e] = (__bf16)v;
}

// ---------------------------------------------------------------------------
// gx GEMM: gx[m,g] = bf16( sum_k xs[m,k] * Wih[g,k] + bih[g] )
// M=32768, N=1536, K in {256,1024}. WG = 256 thr = 8 waves.
// Wave tile 32x32 (2x2 of 16x16), WG tile 128(M) x 64(N).
// Weights & activations are L2-resident -> direct global loads.
// ---------------------------------------------------------------------------
__global__ __launch_bounds__(256)
void gx_gemm_kernel(const __bf16* __restrict__ xs,   // [M, K]
                    const __bf16* __restrict__ w,    // [1536, K]
                    const float*  __restrict__ bih,  // [1536]
                    __bf16* __restrict__ gx,         // [M, 1536]
                    int K) {
  const int lane  = threadIdx.x & 31;
  const int wid   = threadIdx.x >> 5;     // 0..7
  const int waveM = wid >> 1;             // 0..3
  const int waveN = wid & 1;              // 0..1
  const int mbase = blockIdx.y * 128 + waveM * 32;
  const int nbase = blockIdx.x * 64  + waveN * 32;
  const int khalf = (lane >> 4) * 16;     // K sub-slice per half-wave
  const int lrow  = lane & 15;

  const __bf16* ap0 = xs + (size_t)(mbase + lrow) * K + khalf;
  const __bf16* ap1 = ap0 + (size_t)16 * K;
  const __bf16* bp0 = w  + (size_t)(nbase + lrow) * K + khalf;
  const __bf16* bp1 = bp0 + (size_t)16 * K;

  v8f acc00 = {}, acc01 = {}, acc10 = {}, acc11 = {};

  #pragma unroll 4
  for (int k0 = 0; k0 < K; k0 += 32) {
    __builtin_prefetch(ap0 + k0 + 128, 0, 0);   // global_prefetch_b8
    v16bf a0 = *(const v16bf*)(ap0 + k0);
    v16bf a1 = *(const v16bf*)(ap1 + k0);
    v16bf b0 = *(const v16bf*)(bp0 + k0);
    v16bf b1 = *(const v16bf*)(bp1 + k0);
    acc00 = wmma_bf16(a0, b0, acc00);
    acc01 = wmma_bf16(a0, b1, acc01);
    acc10 = wmma_bf16(a1, b0, acc10);
    acc11 = wmma_bf16(a1, b1, acc11);
  }

  const int drow = 8 * (lane >> 4);   // D: row = mt*16 + drow + r, col = lane&15
  const int dcol = lane & 15;
  v8f* accs[2][2] = {{&acc00, &acc01}, {&acc10, &acc11}};
  #pragma unroll
  for (int mt = 0; mt < 2; ++mt) {
    #pragma unroll
    for (int nt = 0; nt < 2; ++nt) {
      int   gcol = nbase + nt * 16 + dcol;
      float bias = bih[gcol];
      size_t rbase = (size_t)(mbase + mt * 16 + drow);
      v8f a = *accs[mt][nt];
      #pragma unroll
      for (int r = 0; r < 8; ++r)
        gx[(rbase + r) * G3H + gcol] = (__bf16)(a[r] + bias);
    }
  }
}

// ---------------------------------------------------------------------------
// Persistent GRU scan (one direction). 32 WGs x 192 thr (6 waves).
// WG owns hidden cols [j0, j0+16). Wave wid: gate = wid>>1, mtile = wid&1.
// Per step: gh = h @ Whh^T (K=512, 16 WMMAs/wave) + bhh -> LDS; fuse gates;
// write h ping-pong (f32 + bf16) and concat output; global grid barrier.
// ---------------------------------------------------------------------------
__global__ __launch_bounds__(192)
void gru_scan_kernel(const __bf16* __restrict__ gx,    // [M, 1536] (bih folded)
                     const __bf16* __restrict__ whh,   // [1536, 512]
                     const float*  __restrict__ bhh,   // [1536]
                     float*        hf,                 // [2][32][512]
                     __bf16*       hb,                 // [2][32][512]
                     unsigned int* cnt,                // monotonic barrier
                     float* __restrict__ ycat,         // [M, 1024]
                     int dircol, int reverse) {
  __shared__ float gbuf[3][BB][16];

  const int tid   = threadIdx.x;
  const int lane  = tid & 31;
  const int wid   = tid >> 5;         // 0..5
  const int gate  = wid >> 1;         // 0=r 1=z 2=n
  const int mtile = wid & 1;
  const int j0    = blockIdx.x * 16;
  const int khalf = (lane >> 4) * 16;
  const int lrow  = lane & 15;
  const int drow  = 8 * (lane >> 4);
  const int dcol  = lane & 15;

  const __bf16* bp = whh + (size_t)(gate * HH + j0 + lrow) * HH + khalf;
  const float   bias = bhh[gate * HH + j0 + dcol];

  for (int s = 0; s < LL; ++s) {
    const int l   = reverse ? (LL - 1 - s) : s;
    const int par = s & 1;

    // --- recurrent GEMM slice: gh[32x16] for this gate ---
    const __bf16* ap = hb + (size_t)par * (BB * HH)
                          + (size_t)(mtile * 16 + lrow) * HH + khalf;
    v8f acc = {};
    #pragma unroll
    for (int k0 = 0; k0 < HH; k0 += 32) {
      v16bf a = *(const v16bf*)(ap + k0);
      v16bf b = *(const v16bf*)(bp + k0);
      acc = wmma_bf16(a, b, acc);
    }
    #pragma unroll
    for (int r = 0; r < 8; ++r)
      gbuf[gate][mtile * 16 + drow + r][dcol] = acc[r] + bias;
    __syncthreads();

    // --- fused gate elementwise: 512 elems over 192 threads ---
    const float*  hprev  = hf + (size_t)par * (BB * HH);
    float*        hnext  = hf + (size_t)(par ^ 1) * (BB * HH);
    __bf16*       hbnext = hb + (size_t)(par ^ 1) * (BB * HH);
    for (int e = tid; e < BB * 16; e += 192) {
      int bi = e >> 4;
      int jj = e & 15;
      int j  = j0 + jj;
      size_t gxbase = ((size_t)(bi << 10) + l) * G3H + j;
      float rp = gbuf[0][bi][jj] + (float)gx[gxbase];
      float zp = gbuf[1][bi][jj] + (float)gx[gxbase + HH];
      float hn = gbuf[2][bi][jj];
      float r  = 1.f / (1.f + __expf(-rp));
      float z  = 1.f / (1.f + __expf(-zp));
      float n  = tanhf((float)gx[gxbase + 2 * HH] + r * hn);
      float hp = hprev[bi * HH + j];
      float h2 = (1.f - z) * n + z * hp;
      hnext[bi * HH + j]  = h2;
      hbnext[bi * HH + j] = (__bf16)h2;
      ycat[((size_t)(bi << 10) + l) * H2 + dircol + j] = h2;
    }

    // --- grid barrier (monotonic counter; all 32 WGs resident) ---
    __syncthreads();
    if (tid == 0) {
      __threadfence();
      atomicAdd(cnt, 1u);
      unsigned int target = (unsigned int)(SCAN_WGS * (s + 1));
      while (__hip_atomic_load(cnt, __ATOMIC_ACQUIRE,
                               __HIP_MEMORY_SCOPE_AGENT) < target) {
        __builtin_amdgcn_s_sleep(1);
      }
    }
    __syncthreads();
  }
}

// ---------------------------------------------------------------------------
// Pooling + attention + std + hier combine + LayerNorm + mu/logvar heads.
// One WG (256 thr) per batch element; thread t owns h = t + 256*k, k=0..3.
// ---------------------------------------------------------------------------
__global__ __launch_bounds__(256)
void pool_head_kernel(const float* __restrict__ ycat,   // [32,1024,1024]
                      const float* __restrict__ ln_w, const float* __restrict__ ln_b,
                      const float* __restrict__ w_mu, const float* __restrict__ b_mu,
                      const float* __restrict__ w_lv, const float* __restrict__ b_lv,
                      float* __restrict__ out) {        // mu[32*128] ++ lv[32*128]
  __shared__ float rowmean[LL];
  __shared__ float wsum[8];
  __shared__ float finalv[H2];

  const int b = blockIdx.x;
  const int t = threadIdx.x;
  const int lane = t & 31, wv = t >> 5;
  const float* base = ycat + (size_t)b * LL * H2;

  float sum[4] = {0,0,0,0}, ssq[4] = {0,0,0,0}, attn[4] = {0,0,0,0};
  float mx[4], first[4] = {0,0,0,0}, last[4] = {0,0,0,0};
  #pragma unroll
  for (int k = 0; k < 4; ++k) mx[k] = -3.4e38f;

  // Pass 1: per-h stats + per-row mean (for attention softmax)
  for (int l = 0; l < LL; ++l) {
    const float* row = base + (size_t)l * H2;
    float p = 0.f;
    #pragma unroll
    for (int k = 0; k < 4; ++k) {
      float v = row[t + 256 * k];
      sum[k] += v; ssq[k] += v * v; mx[k] = fmaxf(mx[k], v);
      if (l == 0)      first[k] = v;
      if (l == LL - 1) last[k]  = v;
      p += v;
    }
    for (int off = 16; off > 0; off >>= 1) p += __shfl_down(p, off, 32);
    if (lane == 0) wsum[wv] = p;
    __syncthreads();
    if (t == 0) {
      float s = 0.f;
      for (int w = 0; w < 8; ++w) s += wsum[w];
      rowmean[l] = s * (1.0f / H2);
    }
    __syncthreads();
  }

  // Softmax over rowmean[0..1023] -> store weights back into rowmean
  float lm = -3.4e38f;
  for (int l = t; l < LL; l += 256) lm = fmaxf(lm, rowmean[l]);
  for (int off = 16; off > 0; off >>= 1) lm = fmaxf(lm, __shfl_down(lm, off, 32));
  if (lane == 0) wsum[wv] = lm;
  __syncthreads();
  float gmax = wsum[0];
  for (int w = 1; w < 8; ++w) gmax = fmaxf(gmax, wsum[w]);
  __syncthreads();
  float le = 0.f;
  for (int l = t; l < LL; l += 256) le += __expf(rowmean[l] - gmax);
  for (int off = 16; off > 0; off >>= 1) le += __shfl_down(le, off, 32);
  if (lane == 0) wsum[wv] = le;
  __syncthreads();
  float gsum = 0.f;
  for (int w = 0; w < 8; ++w) gsum += wsum[w];
  float dinv = 1.f / gsum;
  __syncthreads();
  for (int l = t; l < LL; l += 256) rowmean[l] = __expf(rowmean[l] - gmax) * dinv;
  __syncthreads();

  // Pass 2: attention-weighted sum
  for (int l = 0; l < LL; ++l) {
    float w = rowmean[l];
    const float* row = base + (size_t)l * H2;
    #pragma unroll
    for (int k = 0; k < 4; ++k) attn[k] += w * row[t + 256 * k];
  }

  // Combine features
  #pragma unroll
  for (int k = 0; k < 4; ++k) {
    int h = t + 256 * k;
    float avg  = sum[k] * (1.0f / LL);
    float var1 = (ssq[k] - sum[k] * sum[k] * (1.0f / LL)) * (1.0f / (LL - 1));
    float sd   = sqrtf(fmaxf(var1, 0.f));
    float hier = 0.3f * avg + 0.2f * mx[k] + 0.3f * last[k] + 0.2f * first[k];
    finalv[h]  = 0.4f * hier + 0.3f * attn[k] + 0.2f * sd + 0.1f * tanhf(avg);
  }
  __syncthreads();

  // LayerNorm over 1024 dims (ddof=0)
  float ms = 0.f, vs = 0.f;
  #pragma unroll
  for (int k = 0; k < 4; ++k) { float f = finalv[t + 256 * k]; ms += f; vs += f * f; }
  for (int off = 16; off > 0; off >>= 1) {
    ms += __shfl_down(ms, off, 32);
    vs += __shfl_down(vs, off, 32);
  }
  if (lane == 0) { wsum[wv] = ms; }
  __syncthreads();
  float tsum = 0.f; for (int w = 0; w < 8; ++w) tsum += wsum[w];
  __syncthreads();
  if (lane == 0) { wsum[wv] = vs; }
  __syncthreads();
  float tsq = 0.f; for (int w = 0; w < 8; ++w) tsq += wsum[w];
  __syncthreads();
  float mean = tsum * (1.0f / H2);
  float var  = tsq * (1.0f / H2) - mean * mean;
  float rs   = rsqrtf(var + 1e-5f);
  #pragma unroll
  for (int k = 0; k < 4; ++k) {
    int h = t + 256 * k;
    finalv[h] = (finalv[h] - mean) * rs * ln_w[h] + ln_b[h];
  }
  __syncthreads();

  // Heads: threads 0..127 -> mu, 128..255 -> logvar
  {
    int j = t;
    const float* wrow; float bias; float* op;
    if (j < 128) { wrow = w_mu + (size_t)j * H2;        bias = b_mu[j];
                   op = out + (size_t)b * 128 + j; }
    else         { int jj = j - 128; wrow = w_lv + (size_t)jj * H2; bias = b_lv[jj];
                   op = out + (size_t)BB * 128 + (size_t)b * 128 + jj; }
    float d = bias;
    for (int k2 = 0; k2 < H2; ++k2) d += finalv[k2] * wrow[k2];
    *op = d;
  }
}

// ---------------------------------------------------------------------------
// Host orchestration
// ---------------------------------------------------------------------------
extern "C" void kernel_launch(void* const* d_in, const int* in_sizes, int n_in,
                              void* d_out, int out_size, void* d_ws, size_t ws_size,
                              hipStream_t stream) {
  (void)in_sizes; (void)n_in; (void)out_size; (void)ws_size;

  const int*   x    = (const int*)d_in[0];
  const int*   c    = (const int*)d_in[1];
  const float* emb  = (const float*)d_in[2];
  const float* cemb = (const float*)d_in[3];
  const float *Wih[2][2], *Whh[2][2], *bih[2][2], *bhh[2][2];
  int idx = 4;
  for (int l = 0; l < 2; ++l)
    for (int d = 0; d < 2; ++d) {
      Wih[l][d] = (const float*)d_in[idx++];
      Whh[l][d] = (const float*)d_in[idx++];
      bih[l][d] = (const float*)d_in[idx++];
      bhh[l][d] = (const float*)d_in[idx++];
    }
  const float* ln_w = (const float*)d_in[20];
  const float* ln_b = (const float*)d_in[21];
  const float* w_mu = (const float*)d_in[22];
  const float* b_mu = (const float*)d_in[23];
  const float* w_lv = (const float*)d_in[24];
  const float* b_lv = (const float*)d_in[25];

  // --- workspace carve-out (256B aligned) ---
  char* ws = (char*)d_ws;
  size_t cur = 0;
  auto alloc = [&](size_t bytes) -> void* {
    void* p = ws + cur;
    cur = (cur + bytes + 255) & ~(size_t)255;
    return p;
  };
  unsigned int* cnt = (unsigned int*)alloc(256);
  float*  hf = (float*) alloc((size_t)2 * BB * HH * sizeof(float));
  __bf16* hb = (__bf16*)alloc((size_t)2 * BB * HH * sizeof(__bf16));
  size_t hregion = cur;   // [0, hregion) zeroed before each scan
  __bf16 *wih_bf[2][2], *whh_bf[2][2];
  for (int l = 0; l < 2; ++l)
    for (int d = 0; d < 2; ++d) {
      wih_bf[l][d] = (__bf16*)alloc((size_t)G3H * (l ? H2 : 256) * sizeof(__bf16));
      whh_bf[l][d] = (__bf16*)alloc((size_t)G3H * HH * sizeof(__bf16));
    }
  __bf16* xs0  = (__bf16*)alloc((size_t)MM * 256 * sizeof(__bf16));
  __bf16* xs1  = (__bf16*)alloc((size_t)MM * H2  * sizeof(__bf16));
  __bf16* gx   = (__bf16*)alloc((size_t)MM * G3H * sizeof(__bf16));
  float*  ycat = (float*) alloc((size_t)MM * H2  * sizeof(float));

  // --- weight conversion to bf16 ---
  for (int l = 0; l < 2; ++l)
    for (int d = 0; d < 2; ++d) {
      int nih = G3H * (l ? H2 : 256);
      f2bf_kernel<<<(nih + 255) / 256, 256, 0, stream>>>(Wih[l][d], wih_bf[l][d], nih);
      int nhh = G3H * HH;
      f2bf_kernel<<<(nhh + 255) / 256, 256, 0, stream>>>(Whh[l][d], whh_bf[l][d], nhh);
    }

  // --- embedding ---
  embed_kernel<<<MM, 256, 0, stream>>>(x, c, emb, cemb, xs0);

  // --- stacked bidirectional GRU ---
  for (int l = 0; l < 2; ++l) {
    const __bf16* xsl = l ? xs1 : xs0;
    int K = l ? H2 : 256;
    for (int d = 0; d < 2; ++d) {
      dim3 ggrid(G3H / 64, MM / 128);
      gx_gemm_kernel<<<ggrid, 256, 0, stream>>>(xsl, wih_bf[l][d], bih[l][d], gx, K);
      hipMemsetAsync(d_ws, 0, hregion, stream);  // zero h0 + barrier counter
      gru_scan_kernel<<<SCAN_WGS, 192, 0, stream>>>(
          gx, whh_bf[l][d], bhh[l][d], hf, hb, cnt, ycat,
          /*dircol=*/d * HH, /*reverse=*/d);
    }
    if (l == 0) {
      int n = MM * H2;
      f2bf_kernel<<<(n + 255) / 256, 256, 0, stream>>>(ycat, xs1, n);
    }
  }

  // --- pooling + heads ---
  pool_head_kernel<<<BB, 256, 0, stream>>>(ycat, ln_w, ln_b,
                                           w_mu, b_mu, w_lv, b_lv,
                                           (float*)d_out);
}
